// SlidingWindowAttention_81973745811692
// MI455X (gfx1250) — matrix operand order repaired and checked by reading, not compiled, not run on hardware
//
#include <hip/hip_runtime.h>
#include <hip/hip_bf16.h>

typedef __attribute__((ext_vector_type(16))) __bf16        v16bf;
typedef __attribute__((ext_vector_type(8)))  float         v8f;
typedef __attribute__((ext_vector_type(4)))  unsigned int  v4u;
typedef __attribute__((ext_vector_type(4)))  unsigned int  u32x4;
typedef __attribute__((ext_vector_type(8)))  int           i32x8;
typedef __attribute__((ext_vector_type(4)))  int           i32x4;

union Frag16 { v16bf b; v4u u[2]; };

constexpr int Bn  = 2;
constexpr int Sn  = 2048;
constexpr int En  = 2048;
constexpr int Hn  = 16;
constexpr int Dn  = 128;
constexpr int WIN = 256;
constexpr int Mn  = Bn * Sn;   // 4096 GEMM rows
constexpr int Kd  = En;        // 2048 reduction dim

#if defined(__AMDGCN__) && __has_builtin(__builtin_amdgcn_tensor_load_to_lds)
#define USE_TDM 1
#else
#define USE_TDM 0
#endif

__device__ __forceinline__ unsigned short f2bf(float f) {
  unsigned u = __float_as_uint(f);
  u += 0x7fffu + ((u >> 16) & 1u);        // round-to-nearest-even
  return (unsigned short)(u >> 16);
}

// Issue a 2D TDM tile load (bf16 elements): global[tile_d1 rows x tile_d0 cols],
// row stride = stride0 elements, into LDS at lds_addr (tight packing).
__device__ __forceinline__ void tdm_load_2d(unsigned lds_addr,
                                            unsigned long long gaddr,
                                            unsigned tile_d0, unsigned tile_d1,
                                            unsigned long long stride0) {
#if USE_TDM
  u32x4 g0;
  g0.x = 1u;                                                // count=1, user desc
  g0.y = lds_addr;                                          // LDS byte address
  g0.z = (unsigned)(gaddr & 0xffffffffull);                 // global_addr[31:0]
  g0.w = (unsigned)((gaddr >> 32) & 0x01ffffffull) | (2u << 30);  // [56:32], type=2
  i32x8 g1;
  g1[0] = (int)(1u << 16);                                  // data_size=1 -> 2B
  g1[1] = (int)((tile_d0 & 0xffffu) << 16);                 // tensor_dim0[15:0]
  g1[2] = (int)((tile_d0 >> 16) | ((tile_d1 & 0xffffu) << 16));   // d0 hi | d1 lo
  g1[3] = (int)((tile_d1 >> 16) | ((tile_d0 & 0xffffu) << 16));   // d1 hi | tile_dim0
  g1[4] = (int)(tile_d1 & 0xffffu);                         // tile_dim1, tile_dim2=0
  g1[5] = (int)(unsigned)(stride0 & 0xffffffffull);         // dim0_stride[31:0]
  g1[6] = (int)(unsigned)((stride0 >> 32) & 0xffffull);     // dim0_stride[47:32]
  g1[7] = 0;                                                // dim1_stride=0
  i32x4 gz = {0, 0, 0, 0};
#if __clang_major__ >= 23
  i32x8 gz8 = {0, 0, 0, 0, 0, 0, 0, 0};
  __builtin_amdgcn_tensor_load_to_lds(g0, g1, gz, gz, gz8, 0);
#else
  __builtin_amdgcn_tensor_load_to_lds(g0, g1, gz, gz, 0);
#endif
#else
  (void)lds_addr; (void)gaddr; (void)tile_d0; (void)tile_d1; (void)stride0;
#endif
}

// ---------------------------------------------------------------- converts
__global__ __launch_bounds__(256) void k_f32_to_bf16(const float* __restrict__ in,
                                                     unsigned short* __restrict__ out) {
  size_t base = ((size_t)blockIdx.x * 256 + threadIdx.x) * 8;
  const float4* p = (const float4*)(in + base);
  float4 a = p[0], b = p[1];
  v4u o;
  o.x = (unsigned)f2bf(a.x) | ((unsigned)f2bf(a.y) << 16);
  o.y = (unsigned)f2bf(a.z) | ((unsigned)f2bf(a.w) << 16);
  o.z = (unsigned)f2bf(b.x) | ((unsigned)f2bf(b.y) << 16);
  o.w = (unsigned)f2bf(b.z) | ((unsigned)f2bf(b.w) << 16);
  *(v4u*)(out + base) = o;
}

// W [K][N] f32 -> WT [N][K] bf16 (32x32 LDS tile transpose)
__global__ __launch_bounds__(256) void k_wtrans(const float* __restrict__ Wm,
                                                unsigned short* __restrict__ WT) {
  __shared__ float t[32][33];
  int id = blockIdx.x;
  int k0 = (id & 63) * 32;
  int n0 = (id >> 6) * 32;
  int tid = threadIdx.x;
#pragma unroll
  for (int j = 0; j < 4; ++j) {
    int e = tid + 256 * j, r = e >> 5, c = e & 31;
    t[r][c] = Wm[(size_t)(k0 + r) * En + n0 + c];
  }
  __syncthreads();
#pragma unroll
  for (int j = 0; j < 4; ++j) {
    int e = tid + 256 * j, r = e >> 5, c = e & 31;
    WT[(size_t)(n0 + r) * Kd + k0 + c] = f2bf(t[c][r]);
  }
}

// V bf16 [B,H,S,D] -> VT bf16 [B,H,D,S]
__global__ __launch_bounds__(256) void k_vtrans(const unsigned short* __restrict__ Vb,
                                                unsigned short* __restrict__ VT) {
  __shared__ unsigned short t[32][33];
  int id = blockIdx.x;
  int d0 = (id & 3) * 32;
  int s0 = ((id >> 2) & 63) * 32;
  int bh = id >> 8;
  size_t base = (size_t)bh * Sn * Dn;
  int tid = threadIdx.x;
#pragma unroll
  for (int j = 0; j < 4; ++j) {
    int e = tid + 256 * j, r = e >> 5, c = e & 31;
    t[r][c] = Vb[base + (size_t)(s0 + r) * Dn + d0 + c];
  }
  __syncthreads();
#pragma unroll
  for (int j = 0; j < 4; ++j) {
    int e = tid + 256 * j, r = e >> 5, c = e & 31;
    VT[base + (size_t)(d0 + r) * Sn + s0 + c] = t[c][r];
  }
}

// ---------------------------------------------------------------- bf16 GEMM
// C[M,N] = A[M,K] * BT[N,K]^T + bias ; 64x128 block tile, 8 waves of 32x32.
// TDM path: double-buffered LDS, wave 0 issues tensor_load_to_lds prefetch of
// the next K-slab while all waves run WMMA on the current one.
__global__ __launch_bounds__(256) void k_gemm_bf16(
    const unsigned short* __restrict__ A,    // bf16 [M,K]
    const unsigned short* __restrict__ BT,   // bf16 [N,K]
    const float* __restrict__ bias,          // [N]
    unsigned short* __restrict__ outBhsdBf,  // bf16 [B,H,S,D] or null
    float* __restrict__ outBhsdF,            // f32  [B,H,S,D] or null
    float* __restrict__ outBseF)             // f32  [B,S,E]   or null
{
  const int tid = threadIdx.x;
  const int m0 = blockIdx.y * 64;
  const int n0 = blockIdx.x * 128;
  const int wave = tid >> 5, lane = tid & 31;
  const int lm = lane & 15, lh = lane >> 4;
  const int wm = (wave >> 2) * 32;
  const int wn = (wave & 3) * 32;

  v8f zero = {0, 0, 0, 0, 0, 0, 0, 0};
  v8f acc[2][2] = {{zero, zero}, {zero, zero}};

#if USE_TDM
  __shared__ __align__(16) unsigned short lA[2][64 * 32];
  __shared__ __align__(16) unsigned short lB[2][128 * 32];
  const bool issuer = (wave == 0);
  const unsigned ldsA[2] = {(unsigned)(size_t)&lA[0][0], (unsigned)(size_t)&lA[1][0]};
  const unsigned ldsB[2] = {(unsigned)(size_t)&lB[0][0], (unsigned)(size_t)&lB[1][0]};
  const unsigned long long gA = (unsigned long long)(size_t)(A + (size_t)m0 * Kd);
  const unsigned long long gB = (unsigned long long)(size_t)(BT + (size_t)n0 * Kd);

  if (issuer) {  // prologue: fill buffer 0 with K-slab 0
    tdm_load_2d(ldsA[0], gA, 32u, 64u, (unsigned long long)Kd);
    tdm_load_2d(ldsB[0], gB, 32u, 128u, (unsigned long long)Kd);
  }
  for (int it = 0; it < Kd / 32; ++it) {
    const int cur = it & 1;
    __builtin_amdgcn_s_wait_tensorcnt(0);  // issuer: slab ready; others: no-op
    __syncthreads();
    if (issuer && (it + 1) < Kd / 32) {    // prefetch next slab into other buffer
      unsigned long long off = (unsigned long long)(it + 1) * 64ull;  // 32 elem * 2B
      tdm_load_2d(ldsA[cur ^ 1], gA + off, 32u, 64u, (unsigned long long)Kd);
      tdm_load_2d(ldsB[cur ^ 1], gB + off, 32u, 128u, (unsigned long long)Kd);
    }
    const unsigned short* pA = &lA[cur][0];
    const unsigned short* pB = &lB[cur][0];
    Frag16 fa[2], fb[2];
#pragma unroll
    for (int t = 0; t < 2; ++t) {
      int row = wm + 16 * t + lm;  // A 16x32: chunks at k=8h and k=16+8h
      fa[t].u[0] = *(const v4u*)&pA[row * 32 + 8 * lh];
      fa[t].u[1] = *(const v4u*)&pA[row * 32 + 16 + 8 * lh];
      int col = wn + 16 * t + lm;  // B 32x16: chunks at k=8h and k=16+8h
      fb[t].u[0] = *(const v4u*)&pB[col * 32 + 8 * lh];
      fb[t].u[1] = *(const v4u*)&pB[col * 32 + 16 + 8 * lh];
    }
#pragma unroll
    for (int i = 0; i < 2; ++i)
#pragma unroll
      for (int j = 0; j < 2; ++j)
        acc[i][j] = __builtin_amdgcn_wmma_f32_16x16x32_bf16(
            false, fa[i].b, false, fb[j].b, (short)0, acc[i][j], false, false);
  }
#else
  // Fallback: per-thread staged single-buffer pipeline.
  __shared__ __align__(16) unsigned short lA[64 * 32];
  __shared__ __align__(16) unsigned short lB[128 * 32];
  const int arow = tid >> 2, akc = (tid & 3) * 8;
  const int brow = tid >> 1, bkc = (tid & 1) * 16;
  for (int k0 = 0; k0 < Kd; k0 += 32) {
    *(v4u*)&lA[arow * 32 + akc] = *(const v4u*)&A[(size_t)(m0 + arow) * Kd + k0 + akc];
    *(v4u*)&lB[brow * 32 + bkc] = *(const v4u*)&BT[(size_t)(n0 + brow) * Kd + k0 + bkc];
    *(v4u*)&lB[brow * 32 + bkc + 8] =
        *(const v4u*)&BT[(size_t)(n0 + brow) * Kd + k0 + bkc + 8];
    __syncthreads();
    Frag16 fa[2], fb[2];
#pragma unroll
    for (int t = 0; t < 2; ++t) {
      int row = wm + 16 * t + lm;
      fa[t].u[0] = *(const v4u*)&lA[row * 32 + 8 * lh];
      fa[t].u[1] = *(const v4u*)&lA[row * 32 + 16 + 8 * lh];
      int col = wn + 16 * t + lm;
      fb[t].u[0] = *(const v4u*)&lB[col * 32 + 8 * lh];
      fb[t].u[1] = *(const v4u*)&lB[col * 32 + 16 + 8 * lh];
    }
#pragma unroll
    for (int i = 0; i < 2; ++i)
#pragma unroll
      for (int j = 0; j < 2; ++j)
        acc[i][j] = __builtin_amdgcn_wmma_f32_16x16x32_bf16(
            false, fa[i].b, false, fb[j].b, (short)0, acc[i][j], false, false);
    __syncthreads();
  }
#endif

#pragma unroll
  for (int i = 0; i < 2; ++i)
#pragma unroll
    for (int j = 0; j < 2; ++j) {
      int gn = n0 + wn + 16 * j + lm;
      float bv = bias[gn];
#pragma unroll
      for (int r = 0; r < 8; ++r) {
        int gm = m0 + wm + 16 * i + r + 8 * lh;  // C-layout: row = r + 8*(lane/16)
        float v = acc[i][j][r] + bv;
        if (outBseF) outBseF[(size_t)gm * En + gn] = v;
        if (outBhsdBf || outBhsdF) {
          int bb = gm >> 11, ss = gm & (Sn - 1);
          int hh = gn >> 7, dd = gn & (Dn - 1);
          size_t idx = ((size_t)(bb * Hn + hh) * Sn + ss) * Dn + dd;
          if (outBhsdBf) outBhsdBf[idx] = f2bf(v);
          if (outBhsdF) outBhsdF[idx] = v;
        }
      }
    }
}

// ---------------------------------------------------------------- attention
// One wave per 16-row q tile; flash-style online softmax over 9 k-steps of 32.
__global__ __launch_bounds__(32) void k_attn(
    const unsigned short* __restrict__ Qb,  // bf16 [B,H,S,D]
    const unsigned short* __restrict__ Kb,  // bf16 [B,H,S,D]
    const unsigned short* __restrict__ VT,  // bf16 [B,H,D,S]
    unsigned short* __restrict__ ctx)       // bf16 [B,S,E]
{
  __shared__ __align__(16) unsigned short lS[16 * 32];
  const int idx = blockIdx.x;
  const int qt = idx & 127;
  const int h = (idx >> 7) & 15;
  const int b = idx >> 11;
  const int q0 = qt * 16;
  const int lane = threadIdx.x;
  const int lm = lane & 15, lh = lane >> 4;
  const float scale = 0.08838834764831845f;  // 1/sqrt(128)

  const size_t headBase = (size_t)(b * Hn + h) * Sn * Dn;
  Frag16 qf[4];
  {
    const unsigned short* qrow = Qb + headBase + (size_t)(q0 + lm) * Dn;
#pragma unroll
    for (int c = 0; c < 4; ++c) {
      qf[c].u[0] = *(const v4u*)&qrow[32 * c + 8 * lh];
      qf[c].u[1] = *(const v4u*)&qrow[32 * c + 16 + 8 * lh];
    }
  }

  v8f zero = {0, 0, 0, 0, 0, 0, 0, 0};
  v8f o[8];
#pragma unroll
  for (int c = 0; c < 8; ++c) o[c] = zero;
  float mi[8], li[8];
#pragma unroll
  for (int r = 0; r < 8; ++r) { mi[r] = -1e30f; li[r] = 0.f; }

  for (int step = 0; step < 9; ++step) {
    const int ks = q0 - WIN + 32 * step;
    v8f sc[2];
    sc[0] = zero; sc[1] = zero;
#pragma unroll
    for (int t = 0; t < 2; ++t) {
      int kk = ks + 16 * t + lm;
      int kc = kk < 0 ? 0 : (kk >= Sn ? Sn - 1 : kk);  // OOB rows masked later
      const unsigned short* krow = Kb + headBase + (size_t)kc * Dn;
#pragma unroll
      for (int c = 0; c < 4; ++c) {  // S += Q[16x32] * K^T[32x16]
        Frag16 fk;
        fk.u[0] = *(const v4u*)&krow[32 * c + 8 * lh];
        fk.u[1] = *(const v4u*)&krow[32 * c + 16 + 8 * lh];
        sc[t] = __builtin_amdgcn_wmma_f32_16x16x32_bf16(
            false, qf[c].b, false, fk.b, (short)0, sc[t], false, false);
      }
    }
    // mask + online softmax; stash P (bf16) into LDS for A-layout reload
#pragma unroll
    for (int r = 0; r < 8; ++r) {
      int qpos = q0 + r + 8 * lh;
      float s0 = sc[0][r] * scale;
      float s1 = sc[1][r] * scale;
      int k0p = ks + lm, k1p = ks + 16 + lm;
      bool v0 = (k0p >= 0) && (k0p <= qpos) && (k0p > qpos - WIN);
      bool v1 = (k1p >= 0) && (k1p <= qpos) && (k1p > qpos - WIN);
      s0 = v0 ? s0 : -1e30f;
      s1 = v1 ? s1 : -1e30f;
      float mx = fmaxf(s0, s1);
      mx = fmaxf(mx, __shfl_xor(mx, 1, 16));
      mx = fmaxf(mx, __shfl_xor(mx, 2, 16));
      mx = fmaxf(mx, __shfl_xor(mx, 4, 16));
      mx = fmaxf(mx, __shfl_xor(mx, 8, 16));
      float nm = fmaxf(mi[r], mx);
      float alpha = __expf(mi[r] - nm);
      float p0 = __expf(s0 - nm);
      float p1 = __expf(s1 - nm);
      li[r] = li[r] * alpha + p0 + p1;
      mi[r] = nm;
#pragma unroll
      for (int c = 0; c < 8; ++c) o[c][r] = o[c][r] * alpha;
      int row = r + 8 * lh;
      lS[row * 32 + lm] = f2bf(p0);
      lS[row * 32 + 16 + lm] = f2bf(p1);
    }
    asm volatile("s_wait_dscnt 0" ::: "memory");  // P stores visible wave-wide
    Frag16 fp;  // A-layout reload of P
    fp.u[0] = *(const v4u*)&lS[lm * 32 + 16 * lh];
    fp.u[1] = *(const v4u*)&lS[lm * 32 + 16 * lh + 8];

    int kst = ks + 16 * lh;  // 16-aligned chunk: fully in-range or fully masked
    kst = kst < 0 ? 0 : (kst > Sn - 16 ? Sn - 16 : kst);
#pragma unroll
    for (int c = 0; c < 8; ++c) {  // O += P[16x32] * V[32x16] per d-chunk
      int d = 16 * c + lm;
      const unsigned short* vrow = VT + (size_t)(b * Hn + h) * Dn * Sn + (size_t)d * Sn;
      Frag16 fv;
      fv.u[0] = *(const v4u*)&vrow[kst];
      fv.u[1] = *(const v4u*)&vrow[kst + 8];
      o[c] = __builtin_amdgcn_wmma_f32_16x16x32_bf16(
          false, fp.b, false, fv.b, (short)0, o[c], false, false);
    }
    asm volatile("s_wait_dscnt 0" ::: "memory");  // lS reads done before overwrite
  }

  float inv[8];
#pragma unroll
  for (int r = 0; r < 8; ++r) {
    float l = li[r];
    l += __shfl_xor(l, 1, 16);
    l += __shfl_xor(l, 2, 16);
    l += __shfl_xor(l, 4, 16);
    l += __shfl_xor(l, 8, 16);
    inv[r] = 1.0f / l;
  }
#pragma unroll
  for (int c = 0; c < 8; ++c) {
    int gcol = h * Dn + 16 * c + lm;
#pragma unroll
    for (int r = 0; r < 8; ++r) {
      int grow = b * Sn + q0 + r + 8 * lh;
      ctx[(size_t)grow * En + gcol] = f2bf(o[c][r] * inv[r]);
    }
  }
}

// ---------------------------------------------------------------- launcher
extern "C" void kernel_launch(void* const* d_in, const int* in_sizes, int n_in,
                              void* d_out, int out_size, void* d_ws, size_t ws_size,
                              hipStream_t stream) {
  (void)in_sizes; (void)n_in; (void)out_size; (void)ws_size;
  const float* x  = (const float*)d_in[0];
  const float* Wq = (const float*)d_in[1];
  const float* bq = (const float*)d_in[2];
  const float* Wk = (const float*)d_in[3];
  const float* bk = (const float*)d_in[4];
  const float* Wv = (const float*)d_in[5];
  const float* bv = (const float*)d_in[6];
  const float* Wo = (const float*)d_in[7];
  const float* bo = (const float*)d_in[8];

  float* outp = (float*)d_out;                     // [B,S,E]
  float* kout = outp + (size_t)Bn * Sn * En;       // [B,H,S,D]
  float* vout = kout + (size_t)Bn * Hn * Sn * Dn;  // [B,H,S,D]

  unsigned short* ws = (unsigned short*)d_ws;
  const size_t MN = (size_t)Mn * En;  // 8,388,608
  const size_t WW = (size_t)En * En;  // 4,194,304
  unsigned short* xb   = ws;
  unsigned short* wtq  = xb + MN;
  unsigned short* wtk  = wtq + WW;
  unsigned short* wtv  = wtk + WW;
  unsigned short* wto  = wtv + WW;
  unsigned short* qb   = wto + WW;
  unsigned short* kb   = qb + MN;
  unsigned short* vb   = kb + MN;
  unsigned short* vt   = vb + MN;
  unsigned short* ctxb = vt + MN;

  k_f32_to_bf16<<<dim3((unsigned)(MN / 2048)), dim3(256), 0, stream>>>(x, xb);
  k_wtrans<<<dim3(4096), dim3(256), 0, stream>>>(Wq, wtq);
  k_wtrans<<<dim3(4096), dim3(256), 0, stream>>>(Wk, wtk);
  k_wtrans<<<dim3(4096), dim3(256), 0, stream>>>(Wv, wtv);
  k_wtrans<<<dim3(4096), dim3(256), 0, stream>>>(Wo, wto);

  dim3 gg(En / 128, Mn / 64);
  k_gemm_bf16<<<gg, dim3(256), 0, stream>>>(xb, wtq, bq, qb, nullptr, nullptr);
  k_gemm_bf16<<<gg, dim3(256), 0, stream>>>(xb, wtk, bk, kb, kout, nullptr);
  k_gemm_bf16<<<gg, dim3(256), 0, stream>>>(xb, wtv, bv, vb, vout, nullptr);

  k_vtrans<<<dim3(Bn * Hn * (Sn / 32) * (Dn / 32)), dim3(256), 0, stream>>>(vb, vt);
  k_attn<<<dim3(Bn * Hn * (Sn / 16)), dim3(32), 0, stream>>>(qb, kb, vt, ctxb);
  k_gemm_bf16<<<gg, dim3(256), 0, stream>>>(ctxb, wto, bo, nullptr, nullptr, outp);
}
